// SioConvPSBlock_4896262717934
// MI455X (gfx1250) — compile-verified, matrix operand order checked by reading, not run on hardware
//
#include <hip/hip_runtime.h>
#include <hip/hip_bf16.h>

// ---------------------------------------------------------------------------
// CDNA5 (gfx1250) SioConv block: register-blocked, double-buffered bf16 WMMA
// GEMMs + chunked fp32 log-space scan.
// ---------------------------------------------------------------------------

typedef __attribute__((ext_vector_type(16))) __bf16 v16bf;
typedef __attribute__((ext_vector_type(8)))  float  v8f;

union FragBF {
    uint4 q[2];
    v16bf v;
};

__device__ __forceinline__ unsigned short f32_to_bf16_rne(float f) {
    unsigned int u = __float_as_uint(f);
    unsigned int r = u + 0x7FFFu + ((u >> 16) & 1u);
    return (unsigned short)(r >> 16);
}

__device__ __forceinline__ float silu_f(float v) {
    return v * (1.0f / (1.0f + __expf(-v)));
}

__device__ __forceinline__ float softplus_f(float v) {
    // stable: max(v,0) + log1p(exp(-|v|))
    return fmaxf(v, 0.0f) + log1pf(__expf(-fabsf(v)));
}

__device__ __forceinline__ v8f wmma_bf16(const FragBF& a, const FragBF& b, v8f c) {
    return __builtin_amdgcn_wmma_f32_16x16x32_bf16(
        false, a.v, false, b.v, (short)0, c, false, false);
}

// ---------------------------------------------------------------------------
// Pack f32 weight [K, Dout] row-major into bf16 WMMA-B fragment layout:
// for each (ktile of 32, ntile of 16): 32 lanes x 16 bf16 elems contiguous.
// Element (k, n): kk=k%32, nn=n%16, half=(kk>>3)&1, chunk=kk>>4, p=kk&7
//   lane = nn + 16*half ; elem = chunk*8 + p
// ---------------------------------------------------------------------------
__global__ __launch_bounds__(256) void pack_w(const float* __restrict__ W,
                                              unsigned short* __restrict__ Wp,
                                              int K, int Dout) {
    size_t i = (size_t)blockIdx.x * 256 + threadIdx.x;
    size_t total = (size_t)K * Dout;
    if (i >= total) return;
    int k = (int)(i / Dout);
    int n = (int)(i % Dout);
    int kt = k >> 5, kk = k & 31;
    int nt = n >> 4, nn = n & 15;
    int half = (kk >> 3) & 1;
    int chunk = kk >> 4;
    int p = kk & 7;
    int lane = nn + (half << 4);
    int e = (chunk << 3) + p;
    size_t off = (((size_t)kt * (Dout >> 4) + nt) * 32 + lane) * 16 + e;
    Wp[off] = f32_to_bf16_rne(W[i]);
}

// ---------------------------------------------------------------------------
// RMSNorm: f32 [tokens, D] -> bf16 [tokens, D].  One block per token.
// ---------------------------------------------------------------------------
__global__ __launch_bounds__(256) void rmsnorm_bf16(const float* __restrict__ x,
                                                    const float* __restrict__ g,
                                                    unsigned short* __restrict__ out,
                                                    int D) {
    __shared__ float red[256];
    size_t tok = blockIdx.x;
    const float* xp = x + tok * D;
    float ss = 0.0f;
    for (int i = threadIdx.x; i < D; i += 256) {
        float v = xp[i];
        ss += v * v;
    }
    red[threadIdx.x] = ss;
    __syncthreads();
    for (int s = 128; s > 0; s >>= 1) {
        if ((int)threadIdx.x < s) red[threadIdx.x] += red[threadIdx.x + s];
        __syncthreads();
    }
    float scale = rsqrtf(red[0] / (float)D + 1e-6f);
    for (int i = threadIdx.x; i < D; i += 256) {
        out[tok * D + i] = f32_to_bf16_rne(xp[i] * scale * g[i]);
    }
}

// ---------------------------------------------------------------------------
// Register-blocked, double-buffered WMMA bf16 GEMM.
// Wave tile = 64 (M) x 32 (N): 4 M-tiles x 2 N-tiles, 8 accumulators.
// k-loop unrolled by 2 with two fragment buffer sets: fragments for step k+1
// are in flight while the 8 WMMAs for step k issue.
// MODE 0: plain f32 store
// MODE 1: silu
// MODE 2: (acc+bias)*mul[idx] + add[idx]     (y-gating + residual)
// MODE 3: acc + bias + add[idx]              (FFN residual)
// ---------------------------------------------------------------------------
template <int MODE>
__global__ __launch_bounds__(256) void gemm_wmma(const unsigned short* __restrict__ A,
                                                 const unsigned short* __restrict__ Wp,
                                                 const float* __restrict__ bias,
                                                 const float* __restrict__ mul,
                                                 const float* __restrict__ add,
                                                 float* __restrict__ out,
                                                 int Nrows, int K, int Dout) {
    const int wave = threadIdx.x >> 5;
    const int lane = threadIdx.x & 31;
    const int tilesN = Dout >> 5;  // 32-wide N supertiles
    const long wt = (long)blockIdx.x * 8 + wave;
    const int mst = (int)(wt / tilesN);  // 64-row M supertile
    const int nst = (int)(wt % tilesN);
    if (mst * 64 >= Nrows) return;

    const int nn = lane & 15;
    const int half = lane >> 4;

    const int row0 = mst * 64 + nn;
    const unsigned short* a0 = A + (size_t)row0 * K + half * 8;
    const size_t aMT = (size_t)16 * K;  // elems between M-tiles

    const int nt0 = nst * 2;
    const unsigned short* b0 = Wp + ((size_t)nt0 * 32 + lane) * 16;
    const size_t bstride = (size_t)(Dout >> 4) * 32 * 16;  // elems per k-tile

    auto loadFrags = [&](FragBF (&fa)[4], FragBF (&fb)[2], int kt) {
        const unsigned short* ap = a0 + (size_t)kt * 32;
#pragma unroll
        for (int mi = 0; mi < 4; ++mi) {
            fa[mi].q[0] = *(const uint4*)(ap + mi * aMT);
            fa[mi].q[1] = *(const uint4*)(ap + mi * aMT + 16);
        }
        const unsigned short* bp = b0 + (size_t)kt * bstride;
#pragma unroll
        for (int ni = 0; ni < 2; ++ni) {
            fb[ni].q[0] = *(const uint4*)(bp + ni * 512);
            fb[ni].q[1] = *(const uint4*)(bp + ni * 512 + 8);
        }
        __builtin_prefetch(ap + 8 * 32, 0, 3);
        __builtin_prefetch(bp + 8 * bstride, 0, 3);
    };

    v8f acc[8] = {};
    FragBF fa0[4], fb0[2], fa1[4], fb1[2];
    const int ktiles = K >> 5;  // always even (32 or 128)
    loadFrags(fa0, fb0, 0);
    for (int kt = 0; kt < ktiles; kt += 2) {
        loadFrags(fa1, fb1, kt + 1);
#pragma unroll
        for (int mi = 0; mi < 4; ++mi)
#pragma unroll
            for (int ni = 0; ni < 2; ++ni)
                acc[mi * 2 + ni] = wmma_bf16(fa0[mi], fb0[ni], acc[mi * 2 + ni]);
        if (kt + 2 < ktiles) loadFrags(fa0, fb0, kt + 2);
#pragma unroll
        for (int mi = 0; mi < 4; ++mi)
#pragma unroll
            for (int ni = 0; ni < 2; ++ni)
                acc[mi * 2 + ni] = wmma_bf16(fa1[mi], fb1[ni], acc[mi * 2 + ni]);
    }

#pragma unroll
    for (int ni = 0; ni < 2; ++ni) {
        const int col = (nt0 + ni) * 16 + nn;
        const float bv = bias[col];
#pragma unroll
        for (int mi = 0; mi < 4; ++mi) {
#pragma unroll
            for (int r = 0; r < 8; ++r) {
                const int rr = mst * 64 + mi * 16 + r + half * 8;
                const size_t idx = (size_t)rr * Dout + col;
                float v = acc[mi * 2 + ni][r] + bv;
                if (MODE == 1) v = silu_f(v);
                else if (MODE == 2) v = v * mul[idx] + add[idx];
                else if (MODE == 3) v = v + add[idx];
                out[idx] = v;
            }
        }
    }
}

// ---------------------------------------------------------------------------
// Fused dual GEMM with SiLU gating (FFN up-projection):
//   gated[N,F](bf16) = (A@W1 + b1) * silu(A@W2 + b2)
// Wave tile = 64 (M) x 16 (N) x 2 weight matrices: 8 WMMAs per 32-K step
// sharing the same 4 A fragments; double-buffered like gemm_wmma.
// ---------------------------------------------------------------------------
__global__ __launch_bounds__(256) void gemm_dual_gate(const unsigned short* __restrict__ A,
                                                      const unsigned short* __restrict__ Wp1,
                                                      const unsigned short* __restrict__ Wp2,
                                                      const float* __restrict__ b1,
                                                      const float* __restrict__ b2,
                                                      unsigned short* __restrict__ out,
                                                      int Nrows, int K, int Dout) {
    const int wave = threadIdx.x >> 5;
    const int lane = threadIdx.x & 31;
    const int tilesN = Dout >> 4;
    const long wt = (long)blockIdx.x * 8 + wave;
    const int mst = (int)(wt / tilesN);  // 64-row M supertile
    const int nt = (int)(wt % tilesN);
    if (mst * 64 >= Nrows) return;

    const int nn = lane & 15;
    const int half = lane >> 4;

    const int row0 = mst * 64 + nn;
    const unsigned short* a0 = A + (size_t)row0 * K + half * 8;
    const size_t aMT = (size_t)16 * K;

    const size_t boff = ((size_t)nt * 32 + lane) * 16;
    const unsigned short* b01 = Wp1 + boff;
    const unsigned short* b02 = Wp2 + boff;
    const size_t bstride = (size_t)tilesN * 32 * 16;

    auto loadFrags = [&](FragBF (&fa)[4], FragBF& fb1, FragBF& fb2, int kt) {
        const unsigned short* ap = a0 + (size_t)kt * 32;
#pragma unroll
        for (int mi = 0; mi < 4; ++mi) {
            fa[mi].q[0] = *(const uint4*)(ap + mi * aMT);
            fa[mi].q[1] = *(const uint4*)(ap + mi * aMT + 16);
        }
        const unsigned short* bp1 = b01 + (size_t)kt * bstride;
        const unsigned short* bp2 = b02 + (size_t)kt * bstride;
        fb1.q[0] = *(const uint4*)(bp1);
        fb1.q[1] = *(const uint4*)(bp1 + 8);
        fb2.q[0] = *(const uint4*)(bp2);
        fb2.q[1] = *(const uint4*)(bp2 + 8);
        __builtin_prefetch(ap + 8 * 32, 0, 3);
    };

    v8f acc1[4] = {};
    v8f acc2[4] = {};
    FragBF faA[4], fbA1, fbA2, faB[4], fbB1, fbB2;
    const int ktiles = K >> 5;  // even
    loadFrags(faA, fbA1, fbA2, 0);
    for (int kt = 0; kt < ktiles; kt += 2) {
        loadFrags(faB, fbB1, fbB2, kt + 1);
#pragma unroll
        for (int mi = 0; mi < 4; ++mi) {
            acc1[mi] = wmma_bf16(faA[mi], fbA1, acc1[mi]);
            acc2[mi] = wmma_bf16(faA[mi], fbA2, acc2[mi]);
        }
        if (kt + 2 < ktiles) loadFrags(faA, fbA1, fbA2, kt + 2);
#pragma unroll
        for (int mi = 0; mi < 4; ++mi) {
            acc1[mi] = wmma_bf16(faB[mi], fbB1, acc1[mi]);
            acc2[mi] = wmma_bf16(faB[mi], fbB2, acc2[mi]);
        }
    }

    const int col = nt * 16 + nn;
    const float bv1 = b1[col];
    const float bv2 = b2[col];
#pragma unroll
    for (int mi = 0; mi < 4; ++mi) {
#pragma unroll
        for (int r = 0; r < 8; ++r) {
            const int rr = mst * 64 + mi * 16 + r + half * 8;
            const size_t idx = (size_t)rr * Dout + col;
            float v1 = acc1[mi][r] + bv1;
            float v2 = acc2[mi][r] + bv2;
            out[idx] = f32_to_bf16_rne(v1 * silu_f(v2));
        }
    }
}

// ---------------------------------------------------------------------------
// Chunked selective scan over L per (batch, channel).
// Both cumsum(a) and cumlogsumexp(w) are associative, so split L into
// CH chunks of CS steps:
//   phase1: per-chunk local (sum_a, logsumexp-state m,r) -- fully parallel
//   phase2: tiny exclusive prefix combine over CH chunks per channel
//   phase3: replay each chunk with its prefix state, emit h (f32 + bf16)
//
//   a_t  = -softplus(dt_t)          s_t = prefix + cumsum(a)
//   w_t  = -sp(-z_t) - sp(-dt_t) - s_t
//   h_t  = exp(lse_t + s_t) + exp(s_t)*hidden
// LSE updates are branchless: mn=max(m,w); r=r*e^(m-mn)+e^(w-mn).
// ---------------------------------------------------------------------------
__global__ __launch_bounds__(256) void sio_scan_phase1(const float* __restrict__ dtb,
                                                       const float* __restrict__ zb,
                                                       float* __restrict__ sumA,
                                                       float* __restrict__ sumM,
                                                       float* __restrict__ sumR,
                                                       int BD, int D, int L, int CS) {
    const int CH = L / CS;
    int tid = blockIdx.x * 256 + threadIdx.x;
    if (tid >= BD * CH) return;
    int c = tid / BD;
    int bd = tid % BD;
    int b = bd / D;
    int d = bd % D;
    size_t idx = ((size_t)b * L + (size_t)c * CS) * D + d;
    float s = 0.0f, m = -__builtin_inff(), r = 0.0f;
    for (int t = 0; t < CS; ++t, idx += D) {
        float dt = dtb[idx];
        float zr = zb[idx];
        float a = -softplus_f(dt);          // ln(sigmoid(-dt))
        float ln_da = -softplus_f(-dt);     // ln(sigmoid(dt))
        float z = -softplus_f(-zr);         // ln(sigmoid(zr))
        s += a;
        float w = z + ln_da - s;            // chunk-local s
        float mn = fmaxf(m, w);
        r = r * __expf(m - mn) + __expf(w - mn);
        m = mn;
    }
    sumA[tid] = s;
    sumM[tid] = m;
    sumR[tid] = r;
}

__global__ __launch_bounds__(256) void sio_scan_phase2(const float* __restrict__ sumA,
                                                       const float* __restrict__ sumM,
                                                       const float* __restrict__ sumR,
                                                       float* __restrict__ preS,
                                                       float* __restrict__ preM,
                                                       float* __restrict__ preR,
                                                       int BD, int CH) {
    int bd = blockIdx.x * 256 + threadIdx.x;
    if (bd >= BD) return;
    float S = 0.0f, m = -__builtin_inff(), r = 0.0f;
    for (int c = 0; c < CH; ++c) {
        int i = c * BD + bd;
        preS[i] = S;
        preM[i] = m;
        preR[i] = r;
        float A = sumA[i];
        float mc = sumM[i] - S;   // shift chunk-local lse into global frame
        float rc = sumR[i];
        float mn = fmaxf(m, mc);
        r = r * __expf(m - mn) + rc * __expf(mc - mn);
        m = mn;
        S += A;
    }
}

__global__ __launch_bounds__(256) void sio_scan_phase3(const float* __restrict__ dtb,
                                                       const float* __restrict__ zb,
                                                       const float* __restrict__ hidden,
                                                       const float* __restrict__ preS,
                                                       const float* __restrict__ preM,
                                                       const float* __restrict__ preR,
                                                       float* __restrict__ h_out,
                                                       unsigned short* __restrict__ h_bf,
                                                       int BD, int D, int L, int CS) {
    const int CH = L / CS;
    int tid = blockIdx.x * 256 + threadIdx.x;
    if (tid >= BD * CH) return;
    int c = tid / BD;
    int bd = tid % BD;
    int b = bd / D;
    int d = bd % D;
    float hid = hidden[bd];
    float s = preS[tid];
    float m = preM[tid];
    float r = preR[tid];
    size_t idx = ((size_t)b * L + (size_t)c * CS) * D + d;
    for (int t = 0; t < CS; ++t, idx += D) {
        float dt = dtb[idx];
        float zr = zb[idx];
        float a = -softplus_f(dt);
        float ln_da = -softplus_f(-dt);
        float z = -softplus_f(-zr);
        s += a;
        float w = z + ln_da - s;
        float mn = fmaxf(m, w);
        r = r * __expf(m - mn) + __expf(w - mn);
        m = mn;
        float es = __expf(s);
        float h = __expf(s + m) * r + es * hid;
        h_out[idx] = h;
        h_bf[idx] = f32_to_bf16_rne(h);
    }
}

// ---------------------------------------------------------------------------
// Host-side orchestration
// ---------------------------------------------------------------------------
extern "C" void kernel_launch(void* const* d_in, const int* in_sizes, int n_in,
                              void* d_out, int out_size, void* d_ws, size_t ws_size,
                              hipStream_t stream) {
    (void)in_sizes; (void)n_in; (void)out_size; (void)ws_size;

    const float* x      = (const float*)d_in[0];
    const float* hidden = (const float*)d_in[1];
    const float* w_ln_z = (const float*)d_in[2];
    const float* b_ln_z = (const float*)d_in[3];
    const float* w_dt   = (const float*)d_in[4];
    const float* b_dt   = (const float*)d_in[5];
    const float* w_y    = (const float*)d_in[6];
    const float* b_y    = (const float*)d_in[7];
    const float* w_yg   = (const float*)d_in[8];
    const float* b_yg   = (const float*)d_in[9];
    const float* w_ff   = (const float*)d_in[10];
    const float* b_ff   = (const float*)d_in[11];
    const float* w_ffg  = (const float*)d_in[12];
    const float* b_ffg  = (const float*)d_in[13];
    const float* w_ffo  = (const float*)d_in[14];
    const float* b_ffo  = (const float*)d_in[15];
    const float* g_sio  = (const float*)d_in[16];
    const float* g_ffn  = (const float*)d_in[17];

    const int D = 1024, F = 4096, Bb = 4, L = 4096;
    const int N = Bb * L;       // 16384 tokens
    const int BD = Bb * D;      // 4096 channels
    const int CS = 128;         // scan chunk size
    const int CH = L / CS;      // 32 chunks

    const size_t MB = 1u << 20;
    char* ws = (char*)d_ws;
    // packed bf16 weights (32 MB total)
    unsigned short* wp_dt  = (unsigned short*)(ws + 0 * MB);
    unsigned short* wp_lnz = (unsigned short*)(ws + 2 * MB);
    unsigned short* wp_y   = (unsigned short*)(ws + 4 * MB);
    unsigned short* wp_yg  = (unsigned short*)(ws + 6 * MB);
    unsigned short* wp_ff  = (unsigned short*)(ws + 8 * MB);
    unsigned short* wp_ffg = (unsigned short*)(ws + 16 * MB);
    unsigned short* wp_ffo = (unsigned short*)(ws + 24 * MB);
    // activations
    unsigned short* xn_bf  = (unsigned short*)(ws + 32 * MB);   // [N,D] bf16 (32MB)
    float*          buf1   = (float*)(ws + 64 * MB);            // [N,D] f32 (64MB)
    float*          buf2   = (float*)(ws + 128 * MB);           // [N,D] f32 (64MB)
    unsigned short* h_bf   = (unsigned short*)(ws + 192 * MB);  // [N,D] bf16 (32MB)
    unsigned short* gated  = (unsigned short*)(ws + 224 * MB);  // [N,F] bf16 (128MB)
    // scan chunk summaries (6 x CH*BD floats = 3MB)
    float* scan_ws = (float*)(ws + 352 * MB);
    float* sumA = scan_ws + 0 * (size_t)CH * BD;
    float* sumM = scan_ws + 1 * (size_t)CH * BD;
    float* sumR = scan_ws + 2 * (size_t)CH * BD;
    float* preS = scan_ws + 3 * (size_t)CH * BD;
    float* preM = scan_ws + 4 * (size_t)CH * BD;
    float* preR = scan_ws + 5 * (size_t)CH * BD;

    float* out_x = (float*)d_out;            // [B,L,D]
    float* out_h = out_x + (size_t)N * D;    // [B,L,D]

    // ---- 1. pack all weights into WMMA-B fragment layout ----
    {
        int gDD = (int)(((size_t)D * D + 255) / 256);
        int gDF = (int)(((size_t)D * F + 255) / 256);
        pack_w<<<gDD, 256, 0, stream>>>(w_dt,   wp_dt,  D, D);
        pack_w<<<gDD, 256, 0, stream>>>(w_ln_z, wp_lnz, D, D);
        pack_w<<<gDD, 256, 0, stream>>>(w_y,    wp_y,   D, D);
        pack_w<<<gDD, 256, 0, stream>>>(w_yg,   wp_yg,  D, D);
        pack_w<<<gDF, 256, 0, stream>>>(w_ff,   wp_ff,  D, F);
        pack_w<<<gDF, 256, 0, stream>>>(w_ffg,  wp_ffg, D, F);
        pack_w<<<gDF, 256, 0, stream>>>(w_ffo,  wp_ffo, F, D);
    }

    // gemm_wmma: wave tile 64x32 -> waves = (N/64)*(Dout/32), 8 waves/block
    const int gemmBlocksDD = (N / 64) * (D / 32) / 8;   // 1024
    // gemm_dual_gate: wave tile 64x16 -> waves = (N/64)*(F/16)
    const int dualBlocks   = (N / 64) * (F / 16) / 8;   // 8192

    // ---- 2. RMSNorm(x) -> xn (bf16) ----
    rmsnorm_bf16<<<N, 256, 0, stream>>>(x, g_sio, xn_bf, D);

    // ---- 3/4. dt = xn@w_dt+b ; zraw = xn@w_ln_z+b ----
    gemm_wmma<0><<<gemmBlocksDD, 256, 0, stream>>>(xn_bf, wp_dt,  b_dt,   nullptr, nullptr, buf1, N, D, D);
    gemm_wmma<0><<<gemmBlocksDD, 256, 0, stream>>>(xn_bf, wp_lnz, b_ln_z, nullptr, nullptr, buf2, N, D, D);

    // ---- 5. chunked scan -> h (f32 output + bf16 mirror) ----
    sio_scan_phase1<<<(BD * CH) / 256, 256, 0, stream>>>(buf1, buf2, sumA, sumM, sumR, BD, D, L, CS);
    sio_scan_phase2<<<BD / 256, 256, 0, stream>>>(sumA, sumM, sumR, preS, preM, preR, BD, CH);
    sio_scan_phase3<<<(BD * CH) / 256, 256, 0, stream>>>(buf1, buf2, hidden, preS, preM, preR,
                                                         out_h, h_bf, BD, D, L, CS);

    // ---- 6. gate g = silu(xn@w_yg + b_yg) ----
    gemm_wmma<1><<<gemmBlocksDD, 256, 0, stream>>>(xn_bf, wp_yg, b_yg, nullptr, nullptr, buf1, N, D, D);

    // ---- 7. x1 = (h@w_y + b_y) * g + x ----
    gemm_wmma<2><<<gemmBlocksDD, 256, 0, stream>>>(h_bf, wp_y, b_y, buf1, x, buf2, N, D, D);

    // ---- 8. RMSNorm(x1) -> xn2 (bf16) ----
    rmsnorm_bf16<<<N, 256, 0, stream>>>(buf2, g_ffn, xn_bf, D);

    // ---- 9. gated = (xn2@w_ff+b) * silu(xn2@w_ffg+b) ----
    gemm_dual_gate<<<dualBlocks, 256, 0, stream>>>(xn_bf, wp_ff, wp_ffg, b_ff, b_ffg, gated, N, D, F);

    // ---- 10. out_x = gated@w_ffo + b_ffo + x1 ----
    gemm_wmma<3><<<gemmBlocksDD, 256, 0, stream>>>(gated, wp_ffo, b_ffo, nullptr, buf2, out_x, N, F, D);
}
